// CIFTokenBridge_88819923682053
// MI455X (gfx1250) — matrix operand order repaired and checked by reading, not compiled, not run on hardware
//
#include <hip/hip_runtime.h>
#include <hip/hip_bf16.h>
#include <math.h>

typedef __bf16 bf16;
typedef __attribute__((ext_vector_type(16))) __bf16 v16bf;
typedef __attribute__((ext_vector_type(8)))  __bf16 v8bf;
typedef __attribute__((ext_vector_type(4)))  __bf16 v4bf;
typedef __attribute__((ext_vector_type(8)))  float  v8f;

#define B_   16
#define H_   1024
#define T_   2048
#define KTOK 192

// ---- d_out layout (floats): acoustic [B,K,H] | tok_len [B] | loss [1] ----
#define ACOUSTIC_ELEMS (B_*KTOK*H_)
#define TOK_OFF  ACOUSTIC_ELEMS
#define LOSS_OFF (ACOUSTIC_ELEMS + B_)

// ---- workspace layout (bytes) ----
#define XBF_OFF   ((size_t)0)
#define XBF_BYTES ((size_t)B_*H_*T_*2)      // 64 MiB  bf16 x, [B,H,T]
#define WBF_OFF   (XBF_OFF + XBF_BYTES)
#define WBF_BYTES ((size_t)3*H_*H_*2)       // 6 MiB   bf16 W, [tap,O,I]
#define ACC_OFF   (WBF_OFF + WBF_BYTES)
#define ACC_BYTES ((size_t)B_*T_*4)         // alpha accumulator [B,T] f32
#define ASG_OFF   (ACC_OFF + ACC_BYTES)
#define ASG_BYTES ((size_t)B_*KTOK*T_*2)    // assign^T [B,K,T] bf16

static __device__ __forceinline__ v8f wmma_bf16(v16bf a, v16bf b, v8f c) {
  return __builtin_amdgcn_wmma_f32_16x16x32_bf16(
      /*neg_a=*/false, a, /*neg_b=*/false, b,
      /*c_mod=*/(short)0, c, /*reuse_a=*/false, /*reuse_b=*/false);
}
static __device__ __forceinline__ v16bf cat16(v8bf lo, v8bf hi) {
  v16bf r;
#pragma unroll
  for (int j = 0; j < 8; ++j) { r[j] = lo[j]; r[j + 8] = hi[j]; }
  return r;
}

// ---------------- init: zero alpha accumulator + loss slot ----------------
__global__ void cif_init(float* __restrict__ acc, float* __restrict__ out) {
  int idx = blockIdx.x * blockDim.x + threadIdx.x;
  if (idx < B_ * T_) acc[idx] = 0.f;
  if (idx == 0) out[LOSS_OFF] = 0.f;
}

// ---------------- fp32 -> bf16 conversions ----------------
__global__ void cif_cast_x(const float* __restrict__ enc, bf16* __restrict__ xbf) {
  size_t i4 = (size_t)blockIdx.x * blockDim.x + threadIdx.x;   // 4 elems/thread
  if (i4 >= (size_t)B_ * H_ * T_ / 4) return;
  float4 f = ((const float4*)enc)[i4];
  v4bf v; v[0] = (bf16)f.x; v[1] = (bf16)f.y; v[2] = (bf16)f.z; v[3] = (bf16)f.w;
  ((v4bf*)xbf)[i4] = v;
}
// conv_w [O,I,3] fp32 -> wbf [tap][O][I] bf16
__global__ void cif_cast_w(const float* __restrict__ cw, bf16* __restrict__ wbf) {
  int idx = blockIdx.x * blockDim.x + threadIdx.x;             // o*1024 + i
  if (idx >= H_ * H_) return;
#pragma unroll
  for (int tap = 0; tap < 3; ++tap)
    wbf[(size_t)tap * H_ * H_ + idx] = (bf16)cw[(size_t)idx * 3 + tap];
}

// ---------------- conv-as-GEMM with fused silu/proj epilogue ----------------
// out[b,o,t] = sum_i sum_tap W[o,i,tap] * x[b,i,t+tap-1]; epilogue reduces
// silu(out+bias)*proj_w over o into alpha_acc[b,t] via global f32 atomics.
#define BM 128
#define BN 128
#define BK 32
#define PA 40   // LDS pitch (bf16); 80B = conflict-free 16-lane stride, 16B-aligned
#define PB 40

__global__ __launch_bounds__(256)
void cif_conv_gemm(const bf16* __restrict__ xbf, const bf16* __restrict__ wbf,
                   const float* __restrict__ conv_b, const float* __restrict__ proj_w,
                   float* __restrict__ alpha_acc) {
  __shared__ bf16 As[3][BM * PA];          // 3 taps of W tile [128 x 32]
  __shared__ bf16 BsT[(BN + 2) * PB];      // x tile transposed [130 cols][32 rows]
  __shared__ float bias_s[BM], pw_s[BM];

  const int tid  = threadIdx.x;
  const int lane = tid & 31, wid = tid >> 5;
  const int wm = wid & 1, wn = wid >> 1;   // waves: 2 (M) x 4 (N)
  const int m0 = blockIdx.x * BM;
  const int n0 = blockIdx.y * BN;
  const int b  = blockIdx.z;

  if (tid < BM) { bias_s[tid] = conv_b[m0 + tid]; pw_s[tid] = proj_w[m0 + tid]; }

  v8f vzero = {};
  v8f acc[4][2];
#pragma unroll
  for (int mi = 0; mi < 4; ++mi)
#pragma unroll
    for (int ni = 0; ni < 2; ++ni) acc[mi][ni] = vzero;

  const int arow = tid >> 1, ahalf = tid & 1;   // A tile: 128 rows x 2 halves
  const int brow = tid >> 3, bcg   = tid & 7;   // B tile: 32 rows x 8 col-groups

  for (int i0 = 0; i0 < H_; i0 += BK) {
    // --- stage A: W[m0+r, i0+kl, tap], 16 contiguous bf16 = 2x b128 ---
#pragma unroll
    for (int tap = 0; tap < 3; ++tap) {
      const uint4* g = (const uint4*)(wbf + (size_t)tap * H_ * H_ +
                                      (size_t)(m0 + arow) * H_ + i0 + ahalf * 16);
      uint4* s = (uint4*)(&As[tap][arow * PA + ahalf * 16]);
      s[0] = g[0]; s[1] = g[1];
    }
    // --- stage B: x rows i0..i0+31, cols t = n0-1 .. n0+128 (shared by 3 taps).
    //     Aligned vector loads for t in [n0, n0+128) -> LDS cols 1..128;
    //     only the 2 edge columns use guarded scalar loads. ---
    {
      const bf16* xrow = xbf + ((size_t)b * H_ + i0 + brow) * T_;
      const uint4* g = (const uint4*)(xrow + n0 + bcg * 16);   // 16B-aligned
      uint4 r0 = g[0], r1 = g[1];
      const bf16* rv = (const bf16*)&r0;
      const bf16* rw = (const bf16*)&r1;
#pragma unroll
      for (int j = 0; j < 8; ++j) {
        int c = bcg * 16 + 1 + j;                // col = t - (n0-1)
        BsT[c * PB + brow] = rv[j];
        BsT[(c + 8) * PB + brow] = rw[j];
      }
      if (tid < 64) {
        int r2 = tid >> 1, which = tid & 1;
        int c = which ? (BN + 1) : 0;            // t = n0+128 | n0-1
        int t = n0 - 1 + c;
        const bf16* xr2 = xbf + ((size_t)b * H_ + i0 + r2) * T_;
        BsT[c * PB + r2] = (t >= 0 && t < T_) ? xr2[t] : (bf16)0.0f;
      }
    }
    // prefetch next chunk's tiles into cache while we compute this one
    if (i0 + BK < H_) {
      __builtin_prefetch(wbf + (size_t)(m0 + arow) * H_ + i0 + BK, 0, 3);
      __builtin_prefetch(xbf + ((size_t)b * H_ + i0 + BK + brow) * T_ + n0 + bcg * 16, 0, 3);
    }
    __syncthreads();

#pragma unroll
    for (int tap = 0; tap < 3; ++tap) {
      v16bf af[4];
      const int kba = (lane >> 4) * 8;
#pragma unroll
      for (int mi = 0; mi < 4; ++mi) {
        int row = wm * 64 + mi * 16 + (lane & 15);
        v8bf lo = *(const v8bf*)(&As[tap][row * PA + kba]);
        v8bf hi = *(const v8bf*)(&As[tap][row * PA + 16 + kba]);
        af[mi] = cat16(lo, hi);
      }
      const int kbb = (lane >> 4) * 8;
#pragma unroll
      for (int ni = 0; ni < 2; ++ni) {
        int col = wn * 32 + ni * 16 + (lane & 15) + tap;  // t = n0 + n_local + tap - 1
        v8bf lo = *(const v8bf*)(&BsT[col * PB + kbb]);
        v8bf hi = *(const v8bf*)(&BsT[col * PB + kbb + 16]);
        v16bf bfrag = cat16(lo, hi);
#pragma unroll
        for (int mi = 0; mi < 4; ++mi)
          acc[mi][ni] = wmma_bf16(af[mi], bfrag, acc[mi][ni]);
      }
    }
    __syncthreads();
  }

  // --- epilogue: silu(conv+bias)*proj_w, reduce over the 64 M-rows this wave owns ---
#pragma unroll
  for (int ni = 0; ni < 2; ++ni) {
    int t = n0 + wn * 32 + ni * 16 + (lane & 15);
    float partial = 0.f;
#pragma unroll
    for (int mi = 0; mi < 4; ++mi) {
#pragma unroll
      for (int v = 0; v < 8; ++v) {
        int ml = wm * 64 + mi * 16 + v + 8 * (lane >> 4);
        float val = acc[mi][ni][v] + bias_s[ml];
        float sig = 1.f / (1.f + __expf(-val));
        partial += (val * sig) * pw_s[ml];
      }
    }
    atomicAdd(&alpha_acc[b * T_ + t], partial);
  }
}

// ---------------- per-batch: sigmoid/mask/scan/assign (1 WG per batch) ----------------
__global__ __launch_bounds__(256)
void cif_scan(const float* __restrict__ acc, const int* __restrict__ enc_len,
              const int* __restrict__ tgt_len, const float* __restrict__ proj_b,
              float* __restrict__ out, bf16* __restrict__ asgT) {
  __shared__ float sa_s[T_], cen_s[T_], red[256], denom_s[KTOK];
  const int b = blockIdx.x, tid = threadIdx.x;
  const int len = enc_len[b];
  const int tok = min(max(tgt_len[b], 1), KTOK);
  const float tokf = (float)tok;
  const float pb = proj_b[0];

  float a[8]; float lsum = 0.f;
  const int t0 = tid * 8;
#pragma unroll
  for (int j = 0; j < 8; ++j) {
    int t = t0 + j;
    float x = acc[b * T_ + t] + pb;
    float al = 1.f / (1.f + __expf(-x));
    al = (t < len) ? al : 0.f;
    a[j] = al; lsum += al;
  }
  red[tid] = lsum;
  __syncthreads();
  for (int off = 1; off < 256; off <<= 1) {         // inclusive block scan
    float v = (tid >= off) ? red[tid - off] : 0.f;
    __syncthreads();
    red[tid] += v;
    __syncthreads();
  }
  const float total  = red[255];
  const float prefix = red[tid] - lsum;
  const float scale  = tokf / fmaxf(total, 1e-6f);

  float run = prefix * scale;
  float sa_l[8], cen_l[8];
#pragma unroll
  for (int j = 0; j < 8; ++j) {
    float sa = a[j] * scale;
    run += sa;
    float cen = run - 0.5f * sa;
    sa_l[j] = sa; cen_l[j] = cen;
    sa_s[t0 + j] = sa; cen_s[t0 + j] = cen;
  }
  if (tid == 0) {
    atomicAdd(&out[LOSS_OFF], fabsf(total - tokf) * (1.0f / B_));
    out[TOK_OFF + b] = tokf;
  }
  if (tid < KTOK) denom_s[tid] = 0.f;
  __syncthreads();

  // each frame center touches at most 2 tokens: k in (c-1.5, c+0.5)
#pragma unroll
  for (int j = 0; j < 8; ++j) {
    float c = cen_l[j];
    int k2 = (int)floorf(c + 0.5f);
#pragma unroll
    for (int d2 = 0; d2 < 2; ++d2) {
      int k = k2 - 1 + d2;
      if (k >= 0 && k < tok) {
        float d = fabsf(c - (float)k - 0.5f);
        if (d < 1.f) atomicAdd(&denom_s[k], (1.f - d) * sa_l[j]);
      }
    }
  }
  __syncthreads();

  // dense normalized assign^T [K,T] in bf16 (coalesced writes)
  bf16* dst = asgT + (size_t)b * KTOK * T_;
  for (int idx = tid; idx < KTOK * T_; idx += 256) {
    int k = idx >> 11, t = idx & (T_ - 1);
    float w = 0.f;
    if (k < tok) {
      float d = fabsf(cen_s[t] - (float)k - 0.5f);
      if (d < 1.f) w = (1.f - d) * sa_s[t] / fmaxf(denom_s[k], 1e-6f);
    }
    dst[idx] = (bf16)w;
  }
}

// ---------------- token GEMM: acoustic[b,k,h] = assign^T[k,:] . x[b,h,:] ----------------
// Both operands are K(=t)-contiguous -> WMMA fragments load straight from global.
__global__ __launch_bounds__(256)
void cif_tok_gemm(const bf16* __restrict__ asgT, const bf16* __restrict__ xbf,
                  float* __restrict__ out) {
  const int tid = threadIdx.x, lane = tid & 31, wid = tid >> 5;
  const int mt = blockIdx.x;                        // token tile, 0..11
  const int b  = blockIdx.z;
  const int nbase = blockIdx.y * 512 + wid * 64;    // each wave owns 4 h-tiles

  v8f vzero = {};
  v8f acc[4];
#pragma unroll
  for (int ni = 0; ni < 4; ++ni) acc[ni] = vzero;

  const int m = mt * 16 + (lane & 15);
  const bf16* arow = asgT + ((size_t)(b * KTOK + m)) * T_;
  const int kb8  = (lane >> 4) * 8;
  const int kb16 = (lane >> 4) * 16;

  for (int kk = 0; kk < T_; kk += 32) {
    v8bf alo = *(const v8bf*)(arow + kk + kb8);
    v8bf ahi = *(const v8bf*)(arow + kk + 16 + kb8);
    v16bf af = cat16(alo, ahi);
#pragma unroll
    for (int ni = 0; ni < 4; ++ni) {
      int h = nbase + ni * 16 + (lane & 15);
      const bf16* brow = xbf + ((size_t)b * H_ + h) * T_ + kk + kb8;
      v8bf blo = *(const v8bf*)(brow);
      v8bf bhi = *(const v8bf*)(brow + 16);
      acc[ni] = wmma_bf16(af, cat16(blo, bhi), acc[ni]);
    }
  }
#pragma unroll
  for (int ni = 0; ni < 4; ++ni) {
    int h = nbase + ni * 16 + (lane & 15);
#pragma unroll
    for (int v = 0; v < 8; ++v) {
      int kt = mt * 16 + v + 8 * (lane >> 4);
      out[((size_t)(b * KTOK + kt)) * H_ + h] = acc[ni][v];
    }
  }
}

// ---------------- host launcher ----------------
extern "C" void kernel_launch(void* const* d_in, const int* in_sizes, int n_in,
                              void* d_out, int out_size, void* d_ws, size_t ws_size,
                              hipStream_t stream) {
  (void)in_sizes; (void)n_in; (void)out_size; (void)ws_size;
  const float* enc    = (const float*)d_in[0];
  const int*   enclen = (const int*)d_in[1];
  const int*   tgtlen = (const int*)d_in[2];
  const float* conv_w = (const float*)d_in[3];
  const float* conv_b = (const float*)d_in[4];
  const float* proj_w = (const float*)d_in[5];
  const float* proj_b = (const float*)d_in[6];
  float* out = (float*)d_out;

  char*  ws  = (char*)d_ws;
  bf16*  xbf = (bf16*)(ws + XBF_OFF);
  bf16*  wbf = (bf16*)(ws + WBF_OFF);
  float* acc = (float*)(ws + ACC_OFF);
  bf16*  asg = (bf16*)(ws + ASG_OFF);

  hipLaunchKernelGGL(cif_init, dim3((B_ * T_ + 255) / 256), dim3(256), 0, stream, acc, out);
  hipLaunchKernelGGL(cif_cast_x, dim3((B_ * H_ * T_ / 4 + 255) / 256), dim3(256), 0, stream,
                     enc, xbf);
  hipLaunchKernelGGL(cif_cast_w, dim3((H_ * H_ + 255) / 256), dim3(256), 0, stream,
                     conv_w, wbf);
  hipLaunchKernelGGL(cif_conv_gemm, dim3(H_ / BM, T_ / BN, B_), dim3(256), 0, stream,
                     xbf, wbf, conv_b, proj_w, acc);
  hipLaunchKernelGGL(cif_scan, dim3(B_), dim3(256), 0, stream,
                     acc, enclen, tgtlen, proj_b, out, asg);
  hipLaunchKernelGGL(cif_tok_gemm, dim3(KTOK / 16, 2, B_), dim3(256), 0, stream,
                     asg, xbf, out);
}
